// T5Gemma2Encoder_70308614635872
// MI455X (gfx1250) — matrix operand order repaired and checked by reading, not compiled, not run on hardware
//
#include <hip/hip_runtime.h>
#include <hip/hip_bf16.h>
#include <math.h>

#define B_    2
#define S_    2048
#define D_    2048
#define HQ_   16
#define HKV_  8
#define HD_   128
#define F_    8192
#define L_    2
#define SOFTCAP_ 50.0f
#define SCALE_   0.08838834764831845f  /* 128^-0.5 */
#define EPS_  1e-6f
#define SQRTD_ 45.254833995939045f

// CDNA5 async-to-LDS staging (ISA 15.18.3: GLOBAL_LOAD_ASYNC_TO_LDS_B128,
// tracked by ASYNCcnt).  Set to 0 to fall back to global->VGPR->ds_store.
#define USE_ASYNC 1

typedef __attribute__((ext_vector_type(16))) __bf16 v16bf;
typedef __attribute__((ext_vector_type(8)))  float  v8f;

static __device__ __forceinline__ __bf16 f2bf(float x) { return (__bf16)x; }

static __device__ __forceinline__ v8f zero8() {
  v8f z;
#pragma unroll
  for (int i = 0; i < 8; ++i) z[i] = 0.f;
  return z;
}

// ---- async copy global -> LDS (16B per lane) --------------------------------
static __device__ __forceinline__ void ld128_to_lds(const __bf16* g, __bf16* l) {
#if USE_ASYNC
  unsigned loff = (unsigned)(size_t)(__attribute__((address_space(3))) __bf16*)l;
  asm volatile("global_load_async_to_lds_b128 %0, %1, off"
               :: "v"(loff), "v"((unsigned long long)(size_t)g)
               : "memory");
#else
  *(uint4*)l = *(const uint4*)g;
#endif
}
static __device__ __forceinline__ void async_wait0() {
#if USE_ASYNC
  asm volatile("s_wait_asynccnt 0x0" ::: "memory");
#endif
}
static __device__ __forceinline__ void async_wait3() {
#if USE_ASYNC
  asm volatile("s_wait_asynccnt 0x3" ::: "memory");
#endif
}
static __device__ __forceinline__ void async_wait4() {
#if USE_ASYNC
  asm volatile("s_wait_asynccnt 0x4" ::: "memory");
#endif
}

// A fragment: lane = M row (lane&15); K elems = two contiguous 8-chunks at
// [k0 + 16*(lane>>4)] and [k0 + 16*(lane>>4) + 8] (16 contiguous k, same order as B fragment)
static __device__ __forceinline__ v16bf load_afrag(const __bf16* p0, const __bf16* p1) {
  v16bf r;
#pragma unroll
  for (int j = 0; j < 8; ++j) { r[j] = p0[j]; r[8 + j] = p1[j]; }
  return r;
}
// B fragment: lane = N col (lane&15); K elems = 16 contiguous at [k0 + 16*(lane>>4)]
static __device__ __forceinline__ v16bf load_bfrag(const __bf16* p) {
  v16bf r;
#pragma unroll
  for (int j = 0; j < 16; ++j) r[j] = p[j];
  return r;
}
static __device__ __forceinline__ v8f wmma_bf16(v16bf a, v16bf b, v8f c) {
  return __builtin_amdgcn_wmma_f32_16x16x32_bf16(false, a, false, b, (short)0, c, false, false);
}
static __device__ __forceinline__ float gelu_tanh(float x) {
  return 0.5f * x * (1.f + tanhf(0.7978845608028654f * (x + 0.044715f * x * x * x)));
}

// ---------------------------------------------------------------- utilities
// fp32 W[K][N] -> bf16 Wt[N][K] (LDS-tiled transpose, both sides coalesced)
__global__ __launch_bounds__(256) void k_f2bf_T(const float* __restrict__ W,
                                                __bf16* __restrict__ Wt,
                                                int K, int N) {
  __shared__ float t[32][33];
  int k0 = blockIdx.y * 32, n0 = blockIdx.x * 32;
  int r = threadIdx.x >> 3, c4 = (threadIdx.x & 7) * 4;
#pragma unroll
  for (int j = 0; j < 4; ++j)
    t[r][c4 + j] = W[(size_t)(k0 + r) * N + n0 + c4 + j];
  __syncthreads();
#pragma unroll
  for (int j = 0; j < 4; ++j)
    Wt[(size_t)(n0 + r) * K + k0 + c4 + j] = f2bf(t[c4 + j][r]);
}

__global__ __launch_bounds__(256) void k_embed(const float* __restrict__ emb,
                                               const int* __restrict__ ids,
                                               float* __restrict__ x) {
  int row = blockIdx.x;                 // b*S + s
  int id = ids[row];
  const float* src = emb + (size_t)id * D_;
  float* dst = x + (size_t)row * D_;
#pragma unroll
  for (int j = 0; j < 8; ++j) {
    int c = threadIdx.x * 8 + j;
    dst[c] = src[c] * SQRTD_;
  }
}

__global__ void k_seg(const int* __restrict__ pos, int* __restrict__ seg) {
  int b = blockIdx.x;
  if (threadIdx.x == 0) {
    int s0 = 1;
    seg[b * S_] = 1;
    for (int s = 1; s < S_; ++s) {
      if (pos[b * S_ + s] <= pos[b * S_ + s - 1]) s0 += 1;
      seg[b * S_ + s] = s0;
    }
  }
}

// mode 0: write bf16 to outb; mode 1: outf += norm; mode 2: outf = norm
__global__ __launch_bounds__(256) void k_rmsnorm(const float* __restrict__ in,
                                                 const float* __restrict__ w,
                                                 float* __restrict__ outf,
                                                 __bf16* __restrict__ outb, int mode) {
  int row = blockIdx.x, tid = threadIdx.x;
  const float* x = in + (size_t)row * D_;
  float vals[8]; float ss = 0.f;
#pragma unroll
  for (int j = 0; j < 8; ++j) { vals[j] = x[tid * 8 + j]; ss += vals[j] * vals[j]; }
  __shared__ float red[256];
  red[tid] = ss; __syncthreads();
  for (int s = 128; s > 0; s >>= 1) { if (tid < s) red[tid] += red[tid + s]; __syncthreads(); }
  float f = rsqrtf(red[0] / (float)D_ + EPS_);
#pragma unroll
  for (int j = 0; j < 8; ++j) {
    int c = tid * 8 + j;
    float nv = vals[j] * f * (1.f + w[c]);
    size_t o = (size_t)row * D_ + c;
    if (mode == 0)      outb[o] = f2bf(nv);
    else if (mode == 1) outf[o] += nv;
    else                outf[o] = nv;
  }
}

// per-head RMSNorm (HD=128) + RoPE; writes bf16 [B,H,S,HD]
__global__ __launch_bounds__(128) void k_qknorm_rope(const float* __restrict__ X,
                                                     const float* __restrict__ w,
                                                     const int* __restrict__ pos,
                                                     __bf16* __restrict__ out, int H) {
  int h = blockIdx.x, bs = blockIdx.y, d = threadIdx.x;
  int b = bs / S_, s = bs % S_;
  float v = X[(size_t)bs * (H * HD_) + h * HD_ + d];
  __shared__ float red[128]; __shared__ float xr[128];
  red[d] = v * v; __syncthreads();
  for (int st = 64; st > 0; st >>= 1) { if (d < st) red[d] += red[d + st]; __syncthreads(); }
  float f = rsqrtf(red[0] / (float)HD_ + EPS_);
  float xn = v * f * (1.f + w[d]);
  xr[d] = xn; __syncthreads();
  int i = d & 63;
  float freq = (float)pos[bs] * __powf(10000.f, -(float)(2 * i) / (float)HD_);
  float c = __cosf(freq), sn = __sinf(freq);
  float o = (d < 64) ? (xn * c - xr[d + 64] * sn) : (xn * c + xr[d - 64] * sn);
  out[(((size_t)b * H + h) * S_ + s) * HD_ + d] = f2bf(o);
}

// fp32 V [B*S][HKV*HD] -> bf16 transposed [B][HKV][HD][S]
__global__ __launch_bounds__(128) void k_v_reshapeT(const float* __restrict__ vf,
                                                    __bf16* __restrict__ out) {
  int h = blockIdx.x, bs = blockIdx.y, d = threadIdx.x;
  int b = bs / S_, s = bs % S_;
  out[(((size_t)b * HKV_ + h) * HD_ + d) * S_ + s] =
      f2bf(vf[(size_t)bs * (HKV_ * HD_) + h * HD_ + d]);
}

// ----------------------------------------------------------- generic GEMM
// C[M,N](f32) = A[M,K](bf16,row) x Bt[N,K](bf16,row)^T.  Block tile 128x64,
// BK=32, double-buffered async staging.  8 waves, wave tile 32x32:
// rows 32*(w&3), cols 32*(w>>2); 4 WMMAs per K-step.
__global__ __launch_bounds__(256) void k_gemm_nt(const __bf16* __restrict__ A,
                                                 const __bf16* __restrict__ Bt,
                                                 float* __restrict__ C,
                                                 int M, int N, int K) {
  __shared__ __align__(16) __bf16 As[2][128][40];
  __shared__ __align__(16) __bf16 Bs[2][64][40];
  const int tid = threadIdx.x;
  const int w = tid >> 5, lane = tid & 31;
  const int hv = lane >> 4, ln16 = lane & 15;
  const int m0 = blockIdx.y * 128, n0 = blockIdx.x * 64;
  const int tm = (w & 3) * 32, tn = (w >> 2) * 32;
  const int lr = tid >> 2, lc = (tid & 3) * 8;   // 8-elem chunk coords
  const __bf16* aptr0 = A + (size_t)(m0 + lr) * K + lc;
  const __bf16* aptr1 = A + (size_t)(m0 + 64 + lr) * K + lc;
  const __bf16* bptr  = Bt + (size_t)(n0 + lr) * K + lc;
  v8f c00 = zero8(), c01 = zero8(), c10 = zero8(), c11 = zero8();

  ld128_to_lds(aptr0, &As[0][lr][lc]);
  ld128_to_lds(aptr1, &As[0][64 + lr][lc]);
  ld128_to_lds(bptr, &Bs[0][lr][lc]);
  const int nk = K >> 5;
  for (int t = 0; t < nk - 1; ++t) {
    const int buf = t & 1;
    ld128_to_lds(aptr0 + (t + 1) * 32, &As[buf ^ 1][lr][lc]);
    ld128_to_lds(aptr1 + (t + 1) * 32, &As[buf ^ 1][64 + lr][lc]);
    ld128_to_lds(bptr + (t + 1) * 32, &Bs[buf ^ 1][lr][lc]);
    async_wait3();              // tile t complete (in-order), tile t+1 in flight
    __syncthreads();
    v16bf af0 = load_afrag(&As[buf][tm + ln16][hv * 16], &As[buf][tm + ln16][8 + hv * 16]);
    v16bf af1 = load_afrag(&As[buf][tm + 16 + ln16][hv * 16], &As[buf][tm + 16 + ln16][8 + hv * 16]);
    v16bf b0 = load_bfrag(&Bs[buf][tn + ln16][hv * 16]);
    v16bf b1 = load_bfrag(&Bs[buf][tn + 16 + ln16][hv * 16]);
    c00 = wmma_bf16(af0, b0, c00);
    c01 = wmma_bf16(af0, b1, c01);
    c10 = wmma_bf16(af1, b0, c10);
    c11 = wmma_bf16(af1, b1, c11);
    __syncthreads();
  }
  {
    const int buf = (nk - 1) & 1;
    async_wait0();
    __syncthreads();
    v16bf af0 = load_afrag(&As[buf][tm + ln16][hv * 16], &As[buf][tm + ln16][8 + hv * 16]);
    v16bf af1 = load_afrag(&As[buf][tm + 16 + ln16][hv * 16], &As[buf][tm + 16 + ln16][8 + hv * 16]);
    v16bf b0 = load_bfrag(&Bs[buf][tn + ln16][hv * 16]);
    v16bf b1 = load_bfrag(&Bs[buf][tn + 16 + ln16][hv * 16]);
    c00 = wmma_bf16(af0, b0, c00);
    c01 = wmma_bf16(af0, b1, c01);
    c10 = wmma_bf16(af1, b0, c10);
    c11 = wmma_bf16(af1, b1, c11);
  }
#pragma unroll
  for (int i = 0; i < 8; ++i) {
    int row0 = m0 + tm + hv * 8 + i;
    int row1 = row0 + 16;
    C[(size_t)row0 * N + n0 + tn + ln16] = c00[i];
    C[(size_t)row0 * N + n0 + tn + 16 + ln16] = c01[i];
    C[(size_t)row1 * N + n0 + tn + ln16] = c10[i];
    C[(size_t)row1 * N + n0 + tn + 16 + ln16] = c11[i];
  }
}

// act[M,F](bf16) = gelu(A x Wgu_t[0:F]^T) * (A x Wgu_t[F:2F]^T), Wgu_t [2F][K]
// Block tile 128x64, wave tile 32x32 on gate AND up (8 WMMAs per K-step).
__global__ __launch_bounds__(256) void k_gemm_gateup(const __bf16* __restrict__ A,
                                                     const __bf16* __restrict__ Wt,
                                                     __bf16* __restrict__ act,
                                                     int M, int K) {
  __shared__ __align__(16) __bf16 As[2][128][40];
  __shared__ __align__(16) __bf16 Bg[2][64][40];
  __shared__ __align__(16) __bf16 Bu[2][64][40];
  const int tid = threadIdx.x;
  const int w = tid >> 5, lane = tid & 31;
  const int hv = lane >> 4, ln16 = lane & 15;
  const int m0 = blockIdx.y * 128, n0 = blockIdx.x * 64;
  const int tm = (w & 3) * 32, tn = (w >> 2) * 32;
  const int lr = tid >> 2, lc = (tid & 3) * 8;
  const __bf16* aptr0 = A + (size_t)(m0 + lr) * K + lc;
  const __bf16* aptr1 = A + (size_t)(m0 + 64 + lr) * K + lc;
  const __bf16* gptr = Wt + (size_t)(n0 + lr) * K + lc;
  const __bf16* uptr = Wt + (size_t)(F_ + n0 + lr) * K + lc;
  v8f g00 = zero8(), g01 = zero8(), g10 = zero8(), g11 = zero8();
  v8f u00 = zero8(), u01 = zero8(), u10 = zero8(), u11 = zero8();

  ld128_to_lds(aptr0, &As[0][lr][lc]);
  ld128_to_lds(aptr1, &As[0][64 + lr][lc]);
  ld128_to_lds(gptr, &Bg[0][lr][lc]);
  ld128_to_lds(uptr, &Bu[0][lr][lc]);
  const int nk = K >> 5;
  for (int t = 0; t < nk; ++t) {
    const int buf = t & 1;
    if (t + 1 < nk) {
      ld128_to_lds(aptr0 + (t + 1) * 32, &As[buf ^ 1][lr][lc]);
      ld128_to_lds(aptr1 + (t + 1) * 32, &As[buf ^ 1][64 + lr][lc]);
      ld128_to_lds(gptr + (t + 1) * 32, &Bg[buf ^ 1][lr][lc]);
      ld128_to_lds(uptr + (t + 1) * 32, &Bu[buf ^ 1][lr][lc]);
      async_wait4();
    } else {
      async_wait0();
    }
    __syncthreads();
    v16bf af0 = load_afrag(&As[buf][tm + ln16][hv * 16], &As[buf][tm + ln16][8 + hv * 16]);
    v16bf af1 = load_afrag(&As[buf][tm + 16 + ln16][hv * 16], &As[buf][tm + 16 + ln16][8 + hv * 16]);
    v16bf bg0 = load_bfrag(&Bg[buf][tn + ln16][hv * 16]);
    v16bf bg1 = load_bfrag(&Bg[buf][tn + 16 + ln16][hv * 16]);
    v16bf bu0 = load_bfrag(&Bu[buf][tn + ln16][hv * 16]);
    v16bf bu1 = load_bfrag(&Bu[buf][tn + 16 + ln16][hv * 16]);
    g00 = wmma_bf16(af0, bg0, g00);
    g01 = wmma_bf16(af0, bg1, g01);
    g10 = wmma_bf16(af1, bg0, g10);
    g11 = wmma_bf16(af1, bg1, g11);
    u00 = wmma_bf16(af0, bu0, u00);
    u01 = wmma_bf16(af0, bu1, u01);
    u10 = wmma_bf16(af1, bu0, u10);
    u11 = wmma_bf16(af1, bu1, u11);
    __syncthreads();
  }
#pragma unroll
  for (int i = 0; i < 8; ++i) {
    int row0 = m0 + tm + hv * 8 + i;
    int row1 = row0 + 16;
    act[(size_t)row0 * F_ + n0 + tn + ln16] = f2bf(gelu_tanh(g00[i]) * u00[i]);
    act[(size_t)row0 * F_ + n0 + tn + 16 + ln16] = f2bf(gelu_tanh(g01[i]) * u01[i]);
    act[(size_t)row1 * F_ + n0 + tn + ln16] = f2bf(gelu_tanh(g10[i]) * u10[i]);
    act[(size_t)row1 * F_ + n0 + tn + 16 + ln16] = f2bf(gelu_tanh(g11[i]) * u11[i]);
  }
}

// ------------------------------------------------------ flash attention
// grid (S/64, B*HQ), 256 thr. 64 queries/block, stream 64-key tiles.
// Q,K: [B,H,S,HD] bf16.  V: [B,HKV,HD,S] bf16 (pre-transposed).
__global__ __launch_bounds__(256) void k_attn(const __bf16* __restrict__ Qg,
                                              const __bf16* __restrict__ Kg,
                                              const __bf16* __restrict__ Vg,
                                              const int* __restrict__ seg,
                                              const int* __restrict__ amask,
                                              __bf16* __restrict__ Outg, int sliding) {
  __shared__ __align__(16) __bf16 Qs[64][HD_ + 8];
  __shared__ __align__(16) __bf16 KV[128 * 72];    // union: Ks[64][136] / Vt[128][72]
  __shared__ __align__(16) __bf16 Ps[64][72];
  __shared__ float Sc[64][65];
  __shared__ float mrow[64], lrow[64], arow[64];
  __shared__ float pm[64][4], psum[64][4];
  __bf16 (*Ks)[HD_ + 8] = (__bf16 (*)[HD_ + 8])KV;
  __bf16 (*Vt)[72] = (__bf16 (*)[72])KV;

  const int tid = threadIdx.x;
  const int w = tid >> 5, lane = tid & 31;
  const int hv = lane >> 4, ln16 = lane & 15;
  const int b = blockIdx.y / HQ_, h = blockIdx.y % HQ_, hk = h >> 1;
  const int q0 = blockIdx.x * 64;
  const int tm = (w & 3) * 16;
  const int stn = (w >> 2) * 32;   // score col base
  const int otn = (w >> 2) * 64;   // output col base

  const __bf16* qbase = Qg + (((size_t)b * HQ_ + h) * S_ + q0) * HD_;
#pragma unroll
  for (int it = 0; it < 4; ++it) {
    int ch = tid + 256 * it, r = ch >> 4, c = (ch & 15) * 8;
    ld128_to_lds(qbase + (size_t)r * HD_ + c, &Qs[r][c]);
  }
  if (tid < 64) { mrow[tid] = -1e30f; lrow[tid] = 0.f; }
  v8f o0 = zero8(), o1 = zero8(), o2 = zero8(), o3 = zero8();
  async_wait0();
  __syncthreads();

  for (int jb = 0; jb < S_ / 64; ++jb) {
    const int k0 = jb * 64;
    if (sliding && (k0 > q0 + 63 + 256 || k0 + 63 < q0 - 255)) continue;

    const __bf16* kbase = Kg + (((size_t)b * HKV_ + hk) * S_ + k0) * HD_;
#pragma unroll
    for (int it = 0; it < 4; ++it) {
      int ch = tid + 256 * it, r = ch >> 4, c = (ch & 15) * 8;
      ld128_to_lds(kbase + (size_t)r * HD_ + c, &Ks[r][c]);
    }
    async_wait0();
    __syncthreads();

    // scores: S = Q(64x128) x K^T(128x64)
    v8f s0 = zero8(), s1 = zero8();
#pragma unroll
    for (int kk = 0; kk < HD_; kk += 32) {
      v16bf af = load_afrag(&Qs[tm + ln16][kk + hv * 16], &Qs[tm + ln16][kk + 8 + hv * 16]);
      s0 = wmma_bf16(af, load_bfrag(&Ks[stn + ln16][kk + hv * 16]), s0);
      s1 = wmma_bf16(af, load_bfrag(&Ks[stn + 16 + ln16][kk + hv * 16]), s1);
    }
#pragma unroll
    for (int i = 0; i < 8; ++i) {
      int m = tm + hv * 8 + i, qi = q0 + m;
      int segq = seg[b * S_ + qi];
#pragma unroll
      for (int t = 0; t < 2; ++t) {
        int n = stn + t * 16 + ln16, ki = k0 + n;
        float v = (t == 0) ? s0[i] : s1[i];
        v = tanhf(v * (SCALE_ / SOFTCAP_)) * SOFTCAP_;
        bool ok = (amask[b * S_ + ki] != 0) && (seg[b * S_ + ki] == segq);
        if (sliding) { int d = qi - ki; ok = ok && (d >= -256) && (d <= 255); }
        Sc[m][n] = ok ? v : -1e30f;
      }
    }
    __syncthreads();

    // V tile (pre-transposed global; overwrites Ks region -- safe past barrier)
    const __bf16* vbase = Vg + ((size_t)b * HKV_ + hk) * HD_ * S_ + k0;
#pragma unroll
    for (int it = 0; it < 4; ++it) {
      int ch = tid + 256 * it, r = ch >> 3, c = (ch & 7) * 8;   // 128 rows x 64 cols
      ld128_to_lds(vbase + (size_t)r * S_ + c, &Vt[r][c]);
    }

    // online softmax (4 threads per row)
    {
      int r = tid >> 2, qd = tid & 3;
      float mx = -1e30f;
#pragma unroll
      for (int c2 = 0; c2 < 16; ++c2) mx = fmaxf(mx, Sc[r][qd * 16 + c2]);
      pm[r][qd] = mx;
    }
    __syncthreads();
    {
      int r = tid >> 2, qd = tid & 3;
      float mnew = fmaxf(fmaxf(fmaxf(pm[r][0], pm[r][1]), fmaxf(pm[r][2], pm[r][3])), mrow[r]);
      float sacc = 0.f;
#pragma unroll
      for (int c2 = 0; c2 < 16; ++c2) {
        float p = __expf(Sc[r][qd * 16 + c2] - mnew);
        sacc += p;
        Ps[r][qd * 16 + c2] = f2bf(p);
      }
      psum[r][qd] = sacc;
    }
    __syncthreads();
    if (tid < 64) {
      int r = tid;
      float mold = mrow[r];
      float mnew = fmaxf(fmaxf(fmaxf(pm[r][0], pm[r][1]), fmaxf(pm[r][2], pm[r][3])), mold);
      float a = __expf(mold - mnew);
      lrow[r] = lrow[r] * a + psum[r][0] + psum[r][1] + psum[r][2] + psum[r][3];
      mrow[r] = mnew; arow[r] = a;
    }
    async_wait0();
    __syncthreads();

    // rescale accumulators, then O += P(64x64) x V(64x128)
#pragma unroll
    for (int i = 0; i < 8; ++i) {
      float a = arow[tm + hv * 8 + i];
      o0[i] *= a; o1[i] *= a; o2[i] *= a; o3[i] *= a;
    }
#pragma unroll
    for (int kk = 0; kk < 64; kk += 32) {
      v16bf af = load_afrag(&Ps[tm + ln16][kk + hv * 16], &Ps[tm + ln16][kk + 8 + hv * 16]);
      o0 = wmma_bf16(af, load_bfrag(&Vt[otn + 0 * 16 + ln16][kk + hv * 16]), o0);
      o1 = wmma_bf16(af, load_bfrag(&Vt[otn + 1 * 16 + ln16][kk + hv * 16]), o1);
      o2 = wmma_bf16(af, load_bfrag(&Vt[otn + 2 * 16 + ln16][kk + hv * 16]), o2);
      o3 = wmma_bf16(af, load_bfrag(&Vt[otn + 3 * 16 + ln16][kk + hv * 16]), o3);
    }
    __syncthreads();
  }

#pragma unroll
  for (int i = 0; i < 8; ++i) {
    int m = tm + hv * 8 + i;
    float l = lrow[m];
    float inv = (l > 0.f) ? 1.f / l : 0.f;
    size_t ob = ((size_t)b * S_ + q0 + m) * (HQ_ * HD_) + (size_t)h * HD_;
    Outg[ob + otn + 0 * 16 + ln16] = f2bf(o0[i] * inv);
    Outg[ob + otn + 1 * 16 + ln16] = f2bf(o1[i] * inv);
    Outg[ob + otn + 2 * 16 + ln16] = f2bf(o2[i] * inv);
    Outg[ob + otn + 3 * 16 + ln16] = f2bf(o3[i] * inv);
  }
}

// ---------------------------------------------------------------- driver
extern "C" void kernel_launch(void* const* d_in, const int* in_sizes, int n_in,
                              void* d_out, int out_size, void* d_ws, size_t ws_size,
                              hipStream_t stream) {
  (void)in_sizes; (void)n_in; (void)out_size; (void)ws_size;
  const float* embed        = (const float*)d_in[0];
  const float* wq           = (const float*)d_in[1];
  const float* wk           = (const float*)d_in[2];
  const float* wv           = (const float*)d_in[3];
  const float* wo           = (const float*)d_in[4];
  const float* q_norm_w     = (const float*)d_in[5];
  const float* k_norm_w     = (const float*)d_in[6];
  const float* ln_in        = (const float*)d_in[7];
  const float* ln_post_attn = (const float*)d_in[8];
  const float* ln_pre_ffn   = (const float*)d_in[9];
  const float* ln_post_ffn  = (const float*)d_in[10];
  const float* w_gate_up    = (const float*)d_in[11];
  const float* w_down       = (const float*)d_in[12];
  const float* final_norm_w = (const float*)d_in[13];
  const int* input_ids      = (const int*)d_in[14];
  const int* amask          = (const int*)d_in[15];
  const int* pos            = (const int*)d_in[16];

  char* base = (char*)d_ws;
  size_t off = 0;
  auto alloc = [&](size_t bytes) -> char* {
    char* p = base + off;
    off += (bytes + 255) & ~(size_t)255;
    return p;
  };
  const size_t MR = (size_t)B_ * S_;            // 4096 rows
  const size_t NQ = (size_t)L_ * D_ * HQ_ * HD_;
  const size_t NK = (size_t)L_ * D_ * HKV_ * HD_;
  const size_t NGU = (size_t)L_ * D_ * 2 * F_;
  const size_t NDN = (size_t)L_ * F_ * D_;

  __bf16* wq_t  = (__bf16*)alloc(NQ * 2);   // [L][HQ*HD][D]
  __bf16* wk_t  = (__bf16*)alloc(NK * 2);   // [L][HKV*HD][D]
  __bf16* wv_t  = (__bf16*)alloc(NK * 2);
  __bf16* wo_t  = (__bf16*)alloc(NQ * 2);   // [L][D][HQ*HD]
  __bf16* wgu_t = (__bf16*)alloc(NGU * 2);  // [L][2F][D]
  __bf16* wd_t  = (__bf16*)alloc(NDN * 2);  // [L][D][F]
  float*  x    = (float*)alloc(MR * D_ * 4);
  __bf16* hn   = (__bf16*)alloc(MR * D_ * 2);
  float*  qf   = (float*)alloc(MR * HQ_ * HD_ * 4);
  float*  kf   = (float*)alloc(MR * HKV_ * HD_ * 4);
  float*  vf   = (float*)alloc(MR * HKV_ * HD_ * 4);
  __bf16* qb   = (__bf16*)alloc(MR * HQ_ * HD_ * 2);
  __bf16* kb   = (__bf16*)alloc(MR * HKV_ * HD_ * 2);
  __bf16* vb   = (__bf16*)alloc(MR * HKV_ * HD_ * 2);   // [B][HKV][HD][S]
  __bf16* attn = (__bf16*)alloc(MR * HQ_ * HD_ * 2);
  float*  hf   = (float*)alloc(MR * D_ * 4);
  __bf16* act  = (__bf16*)alloc(MR * F_ * 2);
  int*    seg  = (int*)alloc(MR * 4);

  auto cvtT = [&](const float* s, __bf16* d, int K, int N) {
    k_f2bf_T<<<dim3(N / 32, K / 32), 256, 0, stream>>>(s, d, K, N);
  };
  for (int l = 0; l < L_; ++l) {
    cvtT(wq + (size_t)l * D_ * HQ_ * HD_, wq_t + (size_t)l * D_ * HQ_ * HD_, D_, HQ_ * HD_);
    cvtT(wk + (size_t)l * D_ * HKV_ * HD_, wk_t + (size_t)l * D_ * HKV_ * HD_, D_, HKV_ * HD_);
    cvtT(wv + (size_t)l * D_ * HKV_ * HD_, wv_t + (size_t)l * D_ * HKV_ * HD_, D_, HKV_ * HD_);
    cvtT(wo + (size_t)l * HQ_ * HD_ * D_, wo_t + (size_t)l * HQ_ * HD_ * D_, HQ_ * HD_, D_);
    cvtT(w_gate_up + (size_t)l * D_ * 2 * F_, wgu_t + (size_t)l * D_ * 2 * F_, D_, 2 * F_);
    cvtT(w_down + (size_t)l * F_ * D_, wd_t + (size_t)l * F_ * D_, F_, D_);
  }

  k_seg<<<B_, 32, 0, stream>>>(pos, seg);
  k_embed<<<(int)MR, 256, 0, stream>>>(embed, input_ids, x);

  auto gemm = [&](const __bf16* A, const __bf16* Bt, float* C, int M, int N, int K) {
    dim3 g(N / 64, M / 128);
    k_gemm_nt<<<g, 256, 0, stream>>>(A, Bt, C, M, N, K);
  };

  for (int l = 0; l < L_; ++l) {
    const __bf16* wq_l = wq_t + (size_t)l * D_ * HQ_ * HD_;
    const __bf16* wk_l = wk_t + (size_t)l * D_ * HKV_ * HD_;
    const __bf16* wv_l = wv_t + (size_t)l * D_ * HKV_ * HD_;
    const __bf16* wo_l = wo_t + (size_t)l * HQ_ * HD_ * D_;
    const __bf16* wgu_l = wgu_t + (size_t)l * D_ * 2 * F_;
    const __bf16* wd_l = wd_t + (size_t)l * F_ * D_;

    // h = rmsnorm(x, ln_in) -> bf16
    k_rmsnorm<<<(int)MR, 256, 0, stream>>>(x, ln_in + l * D_, nullptr, hn, 0);
    // qkv projections
    gemm(hn, wq_l, qf, (int)MR, HQ_ * HD_, D_);
    gemm(hn, wk_l, kf, (int)MR, HKV_ * HD_, D_);
    gemm(hn, wv_l, vf, (int)MR, HKV_ * HD_, D_);
    // per-head norm + rope -> bf16 [B,H,S,HD]; V -> [B,HKV,HD,S]
    k_qknorm_rope<<<dim3(HQ_, (int)MR), 128, 0, stream>>>(qf, q_norm_w + l * HD_, pos, qb, HQ_);
    k_qknorm_rope<<<dim3(HKV_, (int)MR), 128, 0, stream>>>(kf, k_norm_w + l * HD_, pos, kb, HKV_);
    k_v_reshapeT<<<dim3(HKV_, (int)MR), 128, 0, stream>>>(vf, vb);
    // flash attention (layer 0 = sliding, layer 1 = full)
    int sliding = (l == 0) ? 1 : 0;
    k_attn<<<dim3(S_ / 64, B_ * HQ_), 256, 0, stream>>>(qb, kb, vb, seg, amask, attn, sliding);
    // output projection, residual add through post-attn norm
    gemm(attn, wo_l, hf, (int)MR, D_, HQ_ * HD_);
    k_rmsnorm<<<(int)MR, 256, 0, stream>>>(hf, ln_post_attn + l * D_, x, nullptr, 1);
    // MLP
    k_rmsnorm<<<(int)MR, 256, 0, stream>>>(x, ln_pre_ffn + l * D_, nullptr, hn, 0);
    k_gemm_gateup<<<dim3(F_ / 64, (int)MR / 128), 256, 0, stream>>>(hn, wgu_l, act, (int)MR, D_);
    gemm(act, wd_l, hf, (int)MR, D_, F_);
    k_rmsnorm<<<(int)MR, 256, 0, stream>>>(hf, ln_post_ffn + l * D_, x, nullptr, 1);
  }
  // final norm -> d_out (fp32)
  k_rmsnorm<<<(int)MR, 256, 0, stream>>>(x, final_norm_w, (float*)d_out, nullptr, 2);
}
